// DTCSensorGNN_72052371357774
// MI455X (gfx1250) — matrix operand order repaired and checked by reading, not compiled, not run on hardware
//
#include <hip/hip_runtime.h>
#include <hip/hip_bf16.h>
#include <math.h>

#define DD 128
#define HH 4

typedef __attribute__((ext_vector_type(16))) _Float16 v16h;
typedef __attribute__((ext_vector_type(8)))  _Float16 v8h;
typedef __attribute__((ext_vector_type(8)))  float    v8f;

// ---------- device helpers ----------
__device__ __forceinline__ float gelu_erf(float x) {
    return 0.5f * x * (1.0f + erff(x * 0.70710678118654752440f));
}
__device__ __forceinline__ float softplus_(float x) {
    return fmaxf(x, 0.0f) + log1pf(expf(-fabsf(x)));
}
// order-preserving float<->uint encoding for atomic max on floats
__device__ __forceinline__ unsigned encf(float f) {
    unsigned u = __float_as_uint(f);
    return (u & 0x80000000u) ? ~u : (u | 0x80000000u);
}
__device__ __forceinline__ float decf(unsigned u) {
    return (u & 0x80000000u) ? __uint_as_float(u & 0x7FFFFFFFu) : __uint_as_float(~u);
}
#define ENC_NEG_INF 0x007FFFFFu  // encf(-inf)

// block-wide (128 threads = 4 wave32) sum reduction
__device__ __forceinline__ float blocksum128(float v, float* sh) {
#pragma unroll
    for (int o = 16; o >= 1; o >>= 1) v += __shfl_xor(v, o, 32);
    int w = threadIdx.x >> 5;
    if ((threadIdx.x & 31) == 0) sh[w] = v;
    __syncthreads();
    float r = sh[0] + sh[1] + sh[2] + sh[3];
    __syncthreads();
    return r;
}

// ---------- generic fills ----------
__global__ void fill_f32(float* p, float v, long n) {
    long i = (long)blockIdx.x * blockDim.x + threadIdx.x;
    if (i < n) p[i] = v;
}
__global__ void fill_u32(unsigned* p, unsigned v, long n) {
    long i = (long)blockIdx.x * blockDim.x + threadIdx.x;
    if (i < n) p[i] = v;
}

// ---------- WMMA GEMM: C[M,128] = A[M,128](f16) @ W[128,128] (+bias) ----------
// Wt is the transposed weight: Wt[o*128 + k] = W[k*128 + o], stored f16.
// Block = 256 threads (8 waves). Each wave: one 16-row tile x two 16-col tiles
// (A fragment reused across two WMMAs). Block covers 32 rows x 128 cols.
// grid.x = ceil(M/32).
__global__ void __launch_bounds__(256)
gemm128_f16_wmma(const _Float16* __restrict__ A,
                 const _Float16* __restrict__ Wt,
                 const float* __restrict__ bias,
                 float* __restrict__ C, int M)
{
    const int lane = threadIdx.x & 31;
    const int wave = threadIdx.x >> 5;            // 0..7
    const int lrow = lane & 15;
    const int half = lane >> 4;                   // 0 or 1

    const int rt = blockIdx.x * 2 + (wave >> 2);  // 16-row tile index
    const int ct = (wave & 3) * 2;                // first of two 16-col tiles

    int arow = rt * 16 + lrow;
    if (arow >= M) arow = M - 1;                  // clamp loads; stores guarded
    const int col0 = ct * 16 + lrow;
    const int col1 = col0 + 16;

    // A frag (16x32 f16): lane row = lrow; 16 contiguous K starting at 16*half
    const _Float16* ap  = A  + (size_t)arow * DD + half * 16;
    // B frag (32x16 f16): lane col; 16 contiguous K starting at 16*half
    const _Float16* bp0 = Wt + (size_t)col0 * DD + half * 16;
    const _Float16* bp1 = Wt + (size_t)col1 * DD + half * 16;

    v8f acc0 = {}, acc1 = {};
#pragma unroll
    for (int ks = 0; ks < DD; ks += 32) {
        v16h a;
        *((v8h*)&a)     = *(const v8h*)(ap + ks);
        *((v8h*)&a + 1) = *(const v8h*)(ap + ks + 8);
        v16h b0 = *(const v16h*)(bp0 + ks);
        v16h b1 = *(const v16h*)(bp1 + ks);
        acc0 = __builtin_amdgcn_wmma_f32_16x16x32_f16(
            false, a, false, b0, (short)0, acc0, false, false);
        acc1 = __builtin_amdgcn_wmma_f32_16x16x32_f16(
            false, a, false, b1, (short)0, acc1, false, false);
    }
    const float bv0 = bias ? bias[col0] : 0.0f;
    const float bv1 = bias ? bias[col1] : 0.0f;
    const int rbase = rt * 16 + half * 8;
    if (rbase + 8 <= M) {                         // fast path: whole fragment in-range
#pragma unroll
        for (int v = 0; v < 8; ++v) {
            C[(size_t)(rbase + v) * DD + col0] = acc0[v] + bv0;
            C[(size_t)(rbase + v) * DD + col1] = acc1[v] + bv1;
        }
    } else {
#pragma unroll
        for (int v = 0; v < 8; ++v) {
            if (rbase + v < M) {
                C[(size_t)(rbase + v) * DD + col0] = acc0[v] + bv0;
                C[(size_t)(rbase + v) * DD + col1] = acc1[v] + bv1;
            }
        }
    }
}

// f16 transpose of a 128x128 f32 weight: Wt[o,k] = W[k,o]
__global__ void w_transpose_f16(const float* __restrict__ W, _Float16* __restrict__ Wt) {
    int i = blockIdx.x * blockDim.x + threadIdx.x;   // 0..16383
    int o = i >> 7, k = i & 127;
    Wt[i] = (_Float16)W[k * DD + o];
}

// ---------- encoders ----------
__global__ void __launch_bounds__(128)
node_encoder(const float* __restrict__ x, const float* __restrict__ w,
             const float* __restrict__ b, const float* __restrict__ g,
             const float* __restrict__ be, float* __restrict__ h,
             _Float16* __restrict__ h16, int N)
{
    __shared__ float sh[4];
    int n = blockIdx.x, j = threadIdx.x;
    float x0 = x[n*4+0], x1 = x[n*4+1], x2 = x[n*4+2], x3 = x[n*4+3];
    float v = b[j] + x0*w[j] + x1*w[DD+j] + x2*w[2*DD+j] + x3*w[3*DD+j];
    float mean = blocksum128(v, sh) * (1.0f/DD);
    float d = v - mean;
    float var = blocksum128(d*d, sh) * (1.0f/DD);
    float y = gelu_erf(d * rsqrtf(var + 1e-5f) * g[j] + be[j]);
    h[(size_t)n*DD + j] = y;
    h16[(size_t)n*DD + j] = (_Float16)y;
}

__global__ void __launch_bounds__(128)
edge_encoder(const float* __restrict__ ea, const float* __restrict__ w,
             const float* __restrict__ b, const int* __restrict__ dst,
             _Float16* __restrict__ eaug16, float* __restrict__ sume,
             float* __restrict__ deg, int E)
{
    int e = blockIdx.x, j = threadIdx.x;
    float a0 = ea[e*3+0], a1 = ea[e*3+1], a2 = ea[e*3+2];
    float v = b[j] + a0*w[j] + a1*w[DD+j] + a2*w[2*DD+j];
    eaug16[(size_t)e*DD + j] = (_Float16)v;
    int d = dst[e];
    atomicAdd(&sume[(size_t)d*DD + j], v);
    if (j == 0) atomicAdd(&deg[d], 1.0f);
}

__global__ void __launch_bounds__(128)
mean_edge(const float* __restrict__ sume, const float* __restrict__ deg,
          _Float16* __restrict__ eaug16, int E)
{
    int n = blockIdx.x, j = threadIdx.x;
    float m = sume[(size_t)n*DD + j] / fmaxf(deg[n], 1.0f);
    eaug16[((size_t)E + n)*DD + j] = (_Float16)m;
}

// ---------- GATv2 edge passes ----------
// Pass 1: m = LeakyReLU(ea + xl[src] + xr[dst]); alpha[e,h] = <m_h, att_h>; amax = segmax
__global__ void __launch_bounds__(128)
gat_alpha(const float* __restrict__ ea_chunk, const float* __restrict__ xl,
          const float* __restrict__ xr, const int* __restrict__ src,
          const int* __restrict__ dst, const float* __restrict__ att,
          float* __restrict__ alpha, unsigned* __restrict__ amax, int e0, int E)
{
    int le = blockIdx.x, ge = e0 + le;
    int s, d;
    if (ge < E) { s = src[ge]; d = dst[ge]; } else { s = ge - E; d = s; }
    int j = threadIdx.x;
    float m = ea_chunk[(size_t)le*DD + j] + xl[(size_t)s*DD + j] + xr[(size_t)d*DD + j];
    m = (m >= 0.0f) ? m : 0.2f * m;
    float t = m * att[j];                      // att flattened [h*32+c] == [j]
#pragma unroll
    for (int o = 16; o >= 1; o >>= 1) t += __shfl_xor(t, o, 32);
    if ((j & 31) == 0) {
        int h = j >> 5;                        // one wave per head
        alpha[(size_t)ge*HH + h] = t;
        atomicMax(&amax[d*HH + h], encf(t));
    }
}

// Pass 2: ex = exp(alpha - amax[dst]); denom[dst] += ex  (ex overwrites alpha)
__global__ void gat_expdenom(const int* __restrict__ dst, float* __restrict__ alpha,
                             const unsigned* __restrict__ amax, float* __restrict__ denom,
                             int E, int Ea)
{
    long i = (long)blockIdx.x * blockDim.x + threadIdx.x;
    if (i >= (long)Ea * HH) return;
    int e = (int)(i >> 2), h = (int)(i & 3);
    int d = (e < E) ? dst[e] : (e - E);
    float ex = expf(alpha[i] - decf(amax[d*HH + h]));
    alpha[i] = ex;
    atomicAdd(&denom[d*HH + h], ex);
}

// Pass 3: agg[dst] += xl[src] * ex/(denom[dst]+1e-16)
__global__ void __launch_bounds__(128)
gat_aggregate(const int* __restrict__ src, const int* __restrict__ dst,
              const float* __restrict__ xl, const float* __restrict__ alpha,
              const float* __restrict__ denom, float* __restrict__ agg, int E)
{
    int e = blockIdx.x;
    int s, d;
    if (e < E) { s = src[e]; d = dst[e]; } else { s = e - E; d = s; }
    int j = threadIdx.x, h = j >> 5;
    float a = alpha[(size_t)e*HH + h] / (denom[d*HH + h] + 1e-16f);
    atomicAdd(&agg[(size_t)d*DD + j], xl[(size_t)s*DD + j] * a);
}

// h = GELU(LN(agg + conv_b + h))   (residual in-place)
__global__ void __launch_bounds__(128)
layer_finalize(const float* __restrict__ agg, const float* __restrict__ convb,
               const float* __restrict__ g, const float* __restrict__ be,
               float* __restrict__ h, _Float16* __restrict__ h16, int N)
{
    __shared__ float sh[4];
    int n = blockIdx.x, j = threadIdx.x;
    float v = agg[(size_t)n*DD + j] + convb[j] + h[(size_t)n*DD + j];
    float mean = blocksum128(v, sh) * (1.0f/DD);
    float d = v - mean;
    float var = blocksum128(d*d, sh) * (1.0f/DD);
    float y = gelu_erf(d * rsqrtf(var + 1e-5f) * g[j] + be[j]);
    h[(size_t)n*DD + j] = y;
    h16[(size_t)n*DD + j] = (_Float16)y;
}

// ---------- pooling + heads ----------
__global__ void __launch_bounds__(128)
pool_accum(const float* __restrict__ h, const int* __restrict__ batch,
           float* __restrict__ pool, float* __restrict__ cnt, int N)
{
    int n = blockIdx.x, j = threadIdx.x;
    int b = batch[n];
    atomicAdd(&pool[(size_t)b*DD + j], h[(size_t)n*DD + j]);
    if (j == 0) atomicAdd(&cnt[b], 1.0f);
}

__global__ void __launch_bounds__(128)
pool_final(const float* __restrict__ pool, const float* __restrict__ cnt,
           float* __restrict__ comb)
{
    int b = blockIdx.x, j = threadIdx.x;
    comb[(size_t)b*256 + j] = pool[(size_t)b*DD + j] / fmaxf(cnt[b], 1.0f);
}

__global__ void __launch_bounds__(128)
global_encoder(const float* __restrict__ gf, const float* __restrict__ w,
               const float* __restrict__ bb, const float* __restrict__ g,
               const float* __restrict__ be, float* __restrict__ comb)
{
    __shared__ float sh[4];
    int b = blockIdx.x, j = threadIdx.x;
    float v = bb[j] + gf[b*3]*w[j] + gf[b*3+1]*w[DD+j] + gf[b*3+2]*w[2*DD+j];
    float mean = blocksum128(v, sh) * (1.0f/DD);
    float d = v - mean;
    float var = blocksum128(d*d, sh) * (1.0f/DD);
    comb[(size_t)b*256 + DD + j] = gelu_erf(d * rsqrtf(var + 1e-5f) * g[j] + be[j]);
}

// out[b,o] = act(bias[o] + sum_k in[b,k]*W[k,o]);  act: 0=GELU, 1=softplus, 2=none
__global__ void dense_act(const float* __restrict__ in, const float* __restrict__ w,
                          const float* __restrict__ b, float* __restrict__ out,
                          int B, int K, int O, int act)
{
    int i = blockIdx.x * blockDim.x + threadIdx.x;
    if (i >= B * O) return;
    int bb = i / O, o = i % O;
    float acc = b[o];
    for (int k = 0; k < K; ++k) acc += in[(size_t)bb*K + k] * w[(size_t)k*O + o];
    if (act == 0) acc = gelu_erf(acc);
    else if (act == 1) acc = softplus_(acc);
    out[i] = acc;
}

// ======================================================================
extern "C" void kernel_launch(void* const* d_in, const int* in_sizes, int n_in,
                              void* d_out, int out_size, void* d_ws, size_t ws_size,
                              hipStream_t stream)
{
    const float* x          = (const float*)d_in[0];
    const int*   edge_index = (const int*)  d_in[1];
    const float* edge_attr  = (const float*)d_in[2];
    const int*   batch      = (const int*)  d_in[3];
    const float* gfeat      = (const float*)d_in[4];
    const float* ne_w = (const float*)d_in[5],  *ne_b = (const float*)d_in[6];
    const float* ne_g = (const float*)d_in[7],  *ne_be= (const float*)d_in[8];
    const float* ee_w = (const float*)d_in[9],  *ee_b = (const float*)d_in[10];
    const float* llw  = (const float*)d_in[11], *llb  = (const float*)d_in[12];
    const float* lrw  = (const float*)d_in[13], *lrb  = (const float*)d_in[14];
    const float* lew  = (const float*)d_in[15];
    const float* attw = (const float*)d_in[16];
    const float* convb= (const float*)d_in[17];
    const float* nrmg = (const float*)d_in[18], *nrmb = (const float*)d_in[19];
    const float* ge_w = (const float*)d_in[20], *ge_b = (const float*)d_in[21];
    const float* ge_g = (const float*)d_in[22], *ge_be= (const float*)d_in[23];
    const float* p1w = (const float*)d_in[24], *p1b = (const float*)d_in[25];
    const float* p2w = (const float*)d_in[26], *p2b = (const float*)d_in[27];
    const float* p3w = (const float*)d_in[28], *p3b = (const float*)d_in[29];
    const float* u1w = (const float*)d_in[30], *u1b = (const float*)d_in[31];
    const float* u2w = (const float*)d_in[32], *u2b = (const float*)d_in[33];

    const int N  = in_sizes[0] / 4;
    const int E  = in_sizes[2] / 3;
    const int B  = in_sizes[4] / 3;
    const int Ea = E + N;
    const int* src = edge_index;
    const int* dst = edge_index + E;

    // ---- carve workspace (~355 MB for the given sizes) ----
    char* wp = (char*)d_ws;
    auto carve = [&](size_t bytes) -> void* {
        void* r = (void*)wp;
        wp += (bytes + 255) & ~(size_t)255;
        return r;
    };
    float*     h      = (float*)    carve((size_t)N  * DD * 4);
    _Float16*  h16    = (_Float16*) carve((size_t)N  * DD * 2);
    float*     xl     = (float*)    carve((size_t)N  * DD * 4);
    float*     xr     = (float*)    carve((size_t)N  * DD * 4);
    _Float16*  eaug16 = (_Float16*) carve((size_t)Ea * DD * 2);
    const int  CH     = 65536;                               // edge chunk (mult of 32)
    float*     eachnk = (float*)    carve((size_t)CH * DD * 4);
    float*     alpha  = (float*)    carve((size_t)Ea * HH * 4);
    unsigned*  amax   = (unsigned*) carve((size_t)N  * HH * 4);
    float*     denom  = (float*)    carve((size_t)N  * HH * 4);
    float*     agg    = (float*)    carve((size_t)N  * DD * 4);
    float*     sume   = (float*)    carve((size_t)N  * DD * 4);
    float*     deg    = (float*)    carve((size_t)N  * 4);
    _Float16*  wt16   = (_Float16*) carve((size_t)9 * DD * DD * 2); // [ll0..2, lr0..2, le0..2]
    float*     pool   = (float*)    carve((size_t)B * DD * 4);
    float*     cnt    = (float*)    carve((size_t)B * 4);
    float*     comb   = (float*)    carve((size_t)B * 256 * 4);
    float*     buf1   = (float*)    carve((size_t)B * DD * 4);
    float*     buf2   = (float*)    carve((size_t)B * 64 * 4);
    float*     ubuf   = (float*)    carve((size_t)B * 64 * 4);

    auto gridf = [](long n, int bs) { return (unsigned)((n + bs - 1) / bs); };

    // ---- encoders + self-loop mean edge attrs ----
    fill_f32<<<gridf((long)N*DD, 256), 256, 0, stream>>>(sume, 0.0f, (long)N*DD);
    fill_f32<<<gridf(N, 256), 256, 0, stream>>>(deg, 0.0f, N);
    node_encoder<<<N, 128, 0, stream>>>(x, ne_w, ne_b, ne_g, ne_be, h, h16, N);
    edge_encoder<<<E, 128, 0, stream>>>(edge_attr, ee_w, ee_b, dst, eaug16, sume, deg, E);
    mean_edge<<<N, 128, 0, stream>>>(sume, deg, eaug16, E);

    // ---- transpose + f16-convert the 9 GAT weight matrices ----
    for (int i = 0; i < 3; ++i) {
        w_transpose_f16<<<64, 256, 0, stream>>>(llw + (size_t)i*DD*DD, wt16 + (size_t)(i  )*DD*DD);
        w_transpose_f16<<<64, 256, 0, stream>>>(lrw + (size_t)i*DD*DD, wt16 + (size_t)(3+i)*DD*DD);
        w_transpose_f16<<<64, 256, 0, stream>>>(lew + (size_t)i*DD*DD, wt16 + (size_t)(6+i)*DD*DD);
    }

    // ---- 3 GATv2 layers ----
    for (int i = 0; i < 3; ++i) {
        const _Float16* wtl = wt16 + (size_t)(i  )*DD*DD;
        const _Float16* wtr = wt16 + (size_t)(3+i)*DD*DD;
        const _Float16* wte = wt16 + (size_t)(6+i)*DD*DD;

        gemm128_f16_wmma<<<gridf(N,32), 256, 0, stream>>>(h16, wtl, llb + i*DD, xl, N);
        gemm128_f16_wmma<<<gridf(N,32), 256, 0, stream>>>(h16, wtr, lrb + i*DD, xr, N);

        fill_u32<<<gridf((long)N*HH, 256), 256, 0, stream>>>(amax, ENC_NEG_INF, (long)N*HH);
        fill_f32<<<gridf((long)N*HH, 256), 256, 0, stream>>>(denom, 0.0f, (long)N*HH);
        fill_f32<<<gridf((long)N*DD, 256), 256, 0, stream>>>(agg, 0.0f, (long)N*DD);

        for (int c0 = 0; c0 < Ea; c0 += CH) {
            int m = (Ea - c0 < CH) ? (Ea - c0) : CH;
            gemm128_f16_wmma<<<gridf(m,32), 256, 0, stream>>>(
                eaug16 + (size_t)c0*DD, wte, nullptr, eachnk, m);
            gat_alpha<<<m, 128, 0, stream>>>(eachnk, xl, xr, src, dst,
                                             attw + i*DD, alpha, amax, c0, E);
        }
        gat_expdenom<<<gridf((long)Ea*HH, 256), 256, 0, stream>>>(dst, alpha, amax, denom, E, Ea);
        gat_aggregate<<<Ea, 128, 0, stream>>>(src, dst, xl, alpha, denom, agg, E);
        layer_finalize<<<N, 128, 0, stream>>>(agg, convb + i*DD, nrmg + i*DD, nrmb + i*DD,
                                              h, h16, N);
    }

    // ---- global mean pool + global encoder -> combined [B,256] ----
    fill_f32<<<gridf((long)B*DD, 256), 256, 0, stream>>>(pool, 0.0f, (long)B*DD);
    fill_f32<<<gridf(B, 256), 256, 0, stream>>>(cnt, 0.0f, B);
    pool_accum<<<N, 128, 0, stream>>>(h, batch, pool, cnt, N);
    pool_final<<<B, 128, 0, stream>>>(pool, cnt, comb);
    global_encoder<<<B, 128, 0, stream>>>(gfeat, ge_w, ge_b, ge_g, ge_be, comb);

    // ---- heads: predictions then uncertainties (concatenated in d_out) ----
    float* outp = (float*)d_out;
    dense_act<<<gridf((long)B*DD, 256), 256, 0, stream>>>(comb, p1w, p1b, buf1, B, 256, DD, 0);
    dense_act<<<gridf((long)B*64, 256), 256, 0, stream>>>(buf1, p2w, p2b, buf2, B, DD, 64, 0);
    dense_act<<<gridf((long)B*3, 256), 256, 0, stream>>>(buf2, p3w, p3b, outp, B, 64, 3, 1);
    dense_act<<<gridf((long)B*64, 256), 256, 0, stream>>>(comb, u1w, u1b, ubuf, B, 256, 64, 0);
    dense_act<<<gridf((long)B*3, 256), 256, 0, stream>>>(ubuf, u2w, u2b, outp + (size_t)B*3, B, 64, 3, 1);
}
